// LocalAttention_39943195852907
// MI455X (gfx1250) — compile-verified, matrix operand order checked by reading, not run on hardware
//
#include <hip/hip_runtime.h>

typedef __attribute__((ext_vector_type(16))) __bf16 v16bf;
typedef __attribute__((ext_vector_type(8)))  __bf16 v8bf;
typedef __attribute__((ext_vector_type(8)))  float  v8f;

namespace {
constexpr int kH   = 1024;
constexpr int kS   = 2048;
constexpr int kNT  = kH / 16;   // 64 n-tiles of 16 cols
constexpr int kKT  = kH / 32;   // 32 k-tiles of 32 dims
constexpr int kPad = 68;        // LDS row stride (floats) for 64-col chunks
}

// ---------- WMMA helpers ----------

__device__ __forceinline__ v8f wmma3(v16bf ah, v16bf al, v16bf bh, v16bf bl, v8f c) {
  // split-bf16: (Ah+Al)(Bh+Bl) ~= Ah*Bh + Ah*Bl + Al*Bh   (drop lo*lo)
  c = __builtin_amdgcn_wmma_f32_16x16x32_bf16(false, ah, false, bh, (short)0, c, false, false);
  c = __builtin_amdgcn_wmma_f32_16x16x32_bf16(false, ah, false, bl, (short)0, c, false, false);
  c = __builtin_amdgcn_wmma_f32_16x16x32_bf16(false, al, false, bh, (short)0, c, false, false);
  return c;
}

// A fragment: 16 seq rows x 32 h-dims from row-major bf16 x.
// ISA 16-bit A layout: lane M = lane&15, kb = (lane&16)?8:0,
// halfs 0..7 -> K kb..kb+7, halfs 8..15 -> K kb+16..kb+23.
__device__ __forceinline__ v16bf load_a(const __bf16* xm, int batch, int rowBase, int kt) {
  const int lane = threadIdx.x & 31;
  int row = rowBase + (lane & 15);
  row = row < 0 ? 0 : (row > kS - 1 ? kS - 1 : row);   // clamped halo rows get prob 0
  const int kb = (lane & 16) ? 8 : 0;
  const __bf16* p = xm + ((size_t)batch * kS + row) * kH + kt * 32 + kb;
  v8bf a = *(const v8bf*)p;
  v8bf b = *(const v8bf*)(p + 16);
  return __builtin_shufflevector(a, b, 0,1,2,3,4,5,6,7,8,9,10,11,12,13,14,15);
}

// B fragment: pre-packed so each lane reads its 16 halfs contiguously (32B).
__device__ __forceinline__ v16bf load_b(const __bf16* wp, int mat, int kt, int nt) {
  const int lane = threadIdx.x & 31;
  const size_t frag = ((size_t)mat * kKT + kt) * kNT + nt;
  return *(const v16bf*)(wp + (frag * 32 + lane) * 16);
}

// Pass-1 GEMM accumulation, specialized on the wave-uniform Q role so the hot
// loop has no exec-mask manipulation or accumulator phi-copies.
template <bool HASQ>
__device__ __forceinline__ void pass1_accum(const __bf16* __restrict__ xhi,
                                            const __bf16* __restrict__ xlo,
                                            const __bf16* __restrict__ whi,
                                            const __bf16* __restrict__ wlo,
                                            int batch, int rowBase, int ntBase,
                                            v8f accK[2], v8f accQ[2]) {
  #pragma unroll 2
  for (int kt = 0; kt < kKT; ++kt) {
    const v16bf ah = load_a(xhi, batch, rowBase, kt);
    const v16bf al = load_a(xlo, batch, rowBase, kt);
    #pragma unroll
    for (int i = 0; i < 2; ++i) {
      const int nt = ntBase + i;
      const v16bf bkh = load_b(whi, 1, kt, nt);
      const v16bf bkl = load_b(wlo, 1, kt, nt);
      accK[i] = wmma3(ah, al, bkh, bkl, accK[i]);
      if (HASQ) {
        const v16bf bqh = load_b(whi, 0, kt, nt);
        const v16bf bql = load_b(wlo, 0, kt, nt);
        accQ[i] = wmma3(ah, al, bqh, bql, accQ[i]);
      }
    }
  }
}

// ---------- prologue: pack weights into B-fragment layout (bf16 hi/lo) ----------

__global__ void pack_weights(const float* __restrict__ Wq, const float* __restrict__ Wk,
                             const float* __restrict__ Wv,
                             __bf16* __restrict__ whi, __bf16* __restrict__ wlo) {
  const int t = blockIdx.x * blockDim.x + threadIdx.x;    // (mat,kt,nt,lane)
  if (t >= 3 * kKT * kNT * 32) return;
  const int lane = t & 31;
  const int nt   = (t >> 5)  & 63;
  const int kt   = (t >> 11) & 31;
  const int mat  = t >> 16;
  const float* W = (mat == 0) ? Wq : (mat == 1) ? Wk : Wv;
  const int n  = nt * 16 + (lane & 15);
  const int kb = (lane & 16) ? 8 : 0;
  const size_t o = (size_t)t * 16;
  #pragma unroll
  for (int h = 0; h < 16; ++h) {
    const int k = kt * 32 + kb + h + ((h < 8) ? 0 : 8);
    const float w = W[(size_t)k * kH + n];
    const __bf16 hi = (__bf16)w;
    whi[o + h] = hi;
    wlo[o + h] = (__bf16)(w - (float)hi);
  }
}

__global__ void pack_x(const float* __restrict__ x, __bf16* __restrict__ xhi,
                       __bf16* __restrict__ xlo, int n) {
  const int t = blockIdx.x * blockDim.x + threadIdx.x;
  if (t >= n) return;
  const float v = x[t];
  const __bf16 hi = (__bf16)v;
  xhi[t] = hi;
  xlo[t] = (__bf16)(v - (float)hi);
}

// ---------- fused local attention ----------
// Block: batch b, 64 query rows [s0, s0+64). K/V halo range: 96 rows [s0-16, s0+80)
// (16-aligned so Q and K/V share A-fragments). 12 waves = 6 row-tiles x 2 n-halves.
// N processed in 16 chunks of 64 cols staged through LDS.

__global__ __launch_bounds__(384)
void local_attn_wmma(const __bf16* __restrict__ xhi, const __bf16* __restrict__ xlo,
                     const __bf16* __restrict__ whi, const __bf16* __restrict__ wlo,
                     const float* __restrict__ bq, const float* __restrict__ bk,
                     const float* __restrict__ bv, float* __restrict__ out) {
  __shared__ float Qs[64 * kPad];   // 17 KB
  __shared__ float Ks[96 * kPad];   // 26 KB (reused as V in pass 2)
  __shared__ float sc[320];         // 64 rows x 5-wide window scores
  __shared__ float pb[320];         // softmax probs

  const int tid  = threadIdx.x;
  const int lane = tid & 31;
  const int wave = tid >> 5;              // 0..11
  const int rt   = wave >> 1;             // 0..5 : shared A row-tile
  const int nh   = wave & 1;              // n-half of the 64-col chunk
  const int batch = blockIdx.x >> 5;
  const int s0    = (blockIdx.x & 31) * 64;
  const int rowBase = s0 - 16 + rt * 16;
  const bool hasQ = (rt >= 1) && (rt <= 4);   // Q row-tile = rt-1
  const int colLane = lane & 15;
  const int rOff = (lane & 16) ? 8 : 0;       // C/D layout: lanes 16-31 hold M+8
  const v8f vzero = {0.f, 0.f, 0.f, 0.f, 0.f, 0.f, 0.f, 0.f};

  for (int i = tid; i < 320; i += 384) sc[i] = 0.0f;
  __syncthreads();

  // ---------------- pass 1: Q,K projection + windowed scores ----------------
  for (int nc = 0; nc < 16; ++nc) {
    v8f accK[2] = {vzero, vzero};
    v8f accQ[2] = {vzero, vzero};
    const int ntBase = nc * 4 + nh * 2;
    if (hasQ)
      pass1_accum<true>(xhi, xlo, whi, wlo, batch, rowBase, ntBase, accK, accQ);
    else
      pass1_accum<false>(xhi, xlo, whi, wlo, batch, rowBase, ntBase, accK, accQ);

    // epilogue: bias add, stage chunk in LDS
    #pragma unroll
    for (int i = 0; i < 2; ++i) {
      const int nt   = ntBase + i;
      const int colL = (nh * 2 + i) * 16 + colLane;
      const float bK = bk[nt * 16 + colLane];
      #pragma unroll
      for (int v = 0; v < 8; ++v)
        Ks[(rt * 16 + rOff + v) * kPad + colL] = accK[i][v] + bK;
      if (hasQ) {
        const float bQ = bq[nt * 16 + colLane];
        #pragma unroll
        for (int v = 0; v < 8; ++v)
          Qs[((rt - 1) * 16 + rOff + v) * kPad + colL] = accQ[i][v] + bQ;
      }
    }
    __syncthreads();
    // windowed score partials for this 64-col chunk: K local row = m + w + 14
    if (tid < 320) {
      const int m = tid / 5, w = tid % 5;
      const float4* q4 = (const float4*)(Qs + m * kPad);
      const float4* k4 = (const float4*)(Ks + (m + w + 14) * kPad);
      float s = 0.f;
      #pragma unroll
      for (int c = 0; c < 16; ++c) {
        const float4 a = q4[c], b = k4[c];
        s += a.x * b.x + a.y * b.y + a.z * b.z + a.w * b.w;
      }
      sc[tid] += s;
    }
    __syncthreads();
  }

  // ---------------- softmax over the 5-wide window ----------------
  if (tid < 64) {
    const int m = tid, s = s0 + m;
    float v[5];
    float mx = -3.0e38f;
    #pragma unroll
    for (int w = 0; w < 5; ++w) {
      const int j = s + w - 2;
      const float val = (j >= 0 && j < kS) ? sc[m * 5 + w] * 0.03125f : -3.0e38f;
      v[w] = val;
      mx = fmaxf(mx, val);
    }
    float sum = 0.f;
    #pragma unroll
    for (int w = 0; w < 5; ++w) { const float e = __expf(v[w] - mx); sum += e; pb[m * 5 + w] = e; }
    const float inv = 1.0f / sum;
    #pragma unroll
    for (int w = 0; w < 5; ++w) pb[m * 5 + w] *= inv;
  }
  __syncthreads();

  // ---------------- pass 2: V projection + P·V output ----------------
  for (int nc = 0; nc < 16; ++nc) {
    v8f accV[2] = {vzero, vzero};
    const int ntBase = nc * 4 + nh * 2;
    #pragma unroll 2
    for (int kt = 0; kt < kKT; ++kt) {
      const v16bf ah = load_a(xhi, batch, rowBase, kt);
      const v16bf al = load_a(xlo, batch, rowBase, kt);
      #pragma unroll
      for (int i = 0; i < 2; ++i) {
        const int nt = ntBase + i;
        const v16bf bvh = load_b(whi, 2, kt, nt);
        const v16bf bvl = load_b(wlo, 2, kt, nt);
        accV[i] = wmma3(ah, al, bvh, bvl, accV[i]);
      }
    }
    #pragma unroll
    for (int i = 0; i < 2; ++i) {
      const int nt   = ntBase + i;
      const int colL = (nh * 2 + i) * 16 + colLane;
      const float bV = bv[nt * 16 + colLane];
      #pragma unroll
      for (int v = 0; v < 8; ++v)
        Ks[(rt * 16 + rOff + v) * kPad + colL] = accV[i][v] + bV;  // V chunk
    }
    __syncthreads();
    for (int e = tid; e < 64 * 64; e += 384) {
      const int m = e >> 6, c = e & 63;
      const float* pm = pb + m * 5;
      const float* vb = Ks + (m + 14) * kPad + c;
      const float o = pm[0] * vb[0]
                    + pm[1] * vb[kPad]
                    + pm[2] * vb[2 * kPad]
                    + pm[3] * vb[3 * kPad]
                    + pm[4] * vb[4 * kPad];
      out[((size_t)batch * kS + (s0 + m)) * kH + nc * 64 + c] = o;
    }
    __syncthreads();
  }
}

// ---------- launch ----------

extern "C" void kernel_launch(void* const* d_in, const int* in_sizes, int n_in,
                              void* d_out, int out_size, void* d_ws, size_t ws_size,
                              hipStream_t stream) {
  (void)in_sizes; (void)n_in; (void)out_size; (void)ws_size;
  const float* x  = (const float*)d_in[0];
  const float* Wq = (const float*)d_in[1];
  const float* bq = (const float*)d_in[2];
  const float* Wk = (const float*)d_in[3];
  const float* bk = (const float*)d_in[4];
  const float* Wv = (const float*)d_in[5];
  const float* bv = (const float*)d_in[6];
  float* out = (float*)d_out;

  char* ws = (char*)d_ws;
  __bf16* whi = (__bf16*)(ws);                            // 6 MB  packed W hi
  __bf16* wlo = (__bf16*)(ws + (size_t)6  * 1024 * 1024); // 6 MB  packed W lo
  __bf16* xhi = (__bf16*)(ws + (size_t)12 * 1024 * 1024); // 16 MB x hi
  __bf16* xlo = (__bf16*)(ws + (size_t)28 * 1024 * 1024); // 16 MB x lo

  const int wThreads = 3 * kKT * kNT * 32;                // 196608
  pack_weights<<<wThreads / 256, 256, 0, stream>>>(Wq, Wk, Wv, whi, wlo);
  const int nx = 4 * kS * kH;                             // 8.4M elements
  pack_x<<<(nx + 255) / 256, 256, 0, stream>>>(x, xhi, xlo, nx);
  local_attn_wmma<<<4 * (kS / 64), 384, 0, stream>>>(xhi, xlo, whi, wlo, bq, bk, bv, out);
}
